// hard_contrast_loss_43361989820671
// MI455X (gfx1250) — matrix operand order repaired and checked by reading, not compiled, not run on hardware
//
#include <hip/hip_runtime.h>
#include <stdint.h>

// ---------------- problem constants ----------------
#define NCLS 8
#define BATCH 4
#define CH 256
#define HS 64
#define WSRC 64
#define HW 65536            // 256*256
#define NWORDS 2048         // HW/32
#define K35 175
#define K15 75
#define Q_BIG (BATCH*K35)   // 700 samples (TP/TN) per class
#define Q_SML (BATCH*K15)   // 300 samples (FP/FN) per class
#define QP_BIG 704          // padded to 16
#define QP_SML 304
#define PT_TILES 19         // 304/16
#define QT_TILES 44         // 704/16
#define PER_PAIR (NCLS*PT_TILES*QT_TILES)   // 6688
#define NTILES (4*PER_PAIR)                 // 26752

// ---------------- workspace layout (bytes) ----------------
#define OFF_PRED 0ull
#define OFF_IDX  262144ull                       // 16000 ints
#define OFF_MASK 326144ull                       // 4*8*4*2048 u32 = 1 MB
#define OFF_TP   (OFF_MASK + 1048576ull)         // 8*704*256 f16
#define OFF_TN   (OFF_TP + 2883584ull)
#define OFF_FP   (OFF_TN + 2883584ull)           // 8*304*256 f16
#define OFF_FN   (OFF_FP + 1245184ull)
#define OFF_PART (OFF_FN + 1245184ull)           // NTILES floats
#define ZERO_WORDS ((2ull*2883584ull + 2ull*1245184ull + 4ull*NTILES)/4ull)

typedef _Float16 h8   __attribute__((ext_vector_type(8)));
typedef _Float16 v16h __attribute__((ext_vector_type(16)));
typedef float    v8f  __attribute__((ext_vector_type(8)));

struct Keys4 { unsigned a0,a1,b0,b1,c0,c1,d0,d1; };  // kt, kn, kfp, kfn

// ---------------- threefry2x32 (jax-compatible) ----------------
__host__ __device__ inline void threefry2x32(unsigned k0, unsigned k1,
                                             unsigned x0, unsigned x1,
                                             unsigned& o0, unsigned& o1) {
  unsigned ks[3] = { k0, k1, k0 ^ k1 ^ 0x1BD11BDAu };
  x0 += ks[0]; x1 += ks[1];
  const int r0[4] = {13,15,26,6}, r1[4] = {17,29,16,24};
  #pragma unroll
  for (int g = 0; g < 5; ++g) {
    const int* r = (g & 1) ? r1 : r0;
    #pragma unroll
    for (int i = 0; i < 4; ++i) {
      x0 += x1;
      x1 = (x1 << r[i]) | (x1 >> (32 - r[i]));
      x1 ^= x0;
    }
    x0 += ks[(g + 1) % 3];
    x1 += ks[(g + 2) % 3] + (unsigned)(g + 1);
  }
  o0 = x0; o1 = x1;
}

__host__ __device__ inline bool mask_eval(int t, bool op, bool ol) {
  switch (t) {
    case 0:  return  op &&  ol;   // TP
    case 1:  return !op && !ol;   // TN
    case 2:  return  op && !ol;   // FP
    default: return !op &&  ol;   // FN
  }
}

// ---------------- kernel 1: pred = argmax_n logit ----------------
__global__ void k_pred(const float* __restrict__ logit, unsigned char* __restrict__ pred) {
  int i = blockIdx.x * blockDim.x + threadIdx.x;       // b*HW + hw
  if (i >= BATCH * HW) return;
  int b = i >> 16, hw = i & (HW - 1);
  const float* p = logit + (size_t)b * NCLS * HW + hw;
  float best = p[0]; int bi = 0;
  #pragma unroll
  for (int n = 1; n < NCLS; ++n) {
    float v = p[(size_t)n * HW];
    if (v > best) { best = v; bi = n; }
  }
  pred[i] = (unsigned char)bi;
}

// ---------------- kernel 2: pack masks into bitfields via ballot ----------------
// maskbits[((t*NCLS+n)*BATCH+b)*NWORDS + word], bit(lane) = mask at hw=word*32+lane
__global__ void k_mask(const unsigned char* __restrict__ pred,
                       const int* __restrict__ label,
                       unsigned* __restrict__ maskbits) {
  int w    = blockIdx.x * (blockDim.x >> 5) + (threadIdx.x >> 5);  // 0..8191
  int lane = threadIdx.x & 31;
  int b = w >> 11, word = w & (NWORDS - 1);
  int hw = (word << 5) + lane;
  int pv = pred[b * HW + hw];
  int lv = label[b * HW + hw];
  #pragma unroll
  for (int t = 0; t < 4; ++t) {
    #pragma unroll
    for (int n = 0; n < NCLS; ++n) {
      unsigned long long bal = __ballot(mask_eval(t, pv == n, lv == n));
      if (lane == 0) maskbits[((t * NCLS + n) * BATCH + b) * NWORDS + word] = (unsigned)bal;
    }
  }
}

// ---------------- kernel 3: zero matrices + partials ----------------
__global__ void k_zero(unsigned* __restrict__ dst, unsigned long long nwords) {
  unsigned long long i = (unsigned long long)blockIdx.x * blockDim.x + threadIdx.x;
  if (i < nwords) dst[i] = 0u;
}

// ---------------- kernel 4: categorical sampling, integer-only hot loop ----------------
// argmax(logits + gumbel) == lexicographic max of (maskbit, bits>>9, -hw):
// gumbel is monotone in u, u is injective-monotone in (bits>>9) for f32 jax uniform,
// and any unmasked score (>= -4.47) beats any masked score (~ -1e9).
// Packed u64 key: mask<<40 | (bits>>9)<<16 | (65535-hw)  ->  plain max reduce.
// One wave owns TWO samples whose threefry counters pair as (i, i+N/2).
__global__ void k_sample(const unsigned* __restrict__ maskbits,
                         Keys4 keys, int* __restrict__ idxArr) {
  int w    = blockIdx.x * (blockDim.x >> 5) + (threadIdx.x >> 5);
  int lane = threadIdx.x & 31;
  int t, p, K, idxOff; unsigned k0, k1;
  if (w < 2800)      { t = 0; p = w;        K = K35; k0 = keys.a0; k1 = keys.a1; idxOff = 0;     }
  else if (w < 5600) { t = 1; p = w - 2800; K = K35; k0 = keys.b0; k1 = keys.b1; idxOff = 5600;  }
  else if (w < 6800) { t = 2; p = w - 5600; K = K15; k0 = keys.c0; k1 = keys.c1; idxOff = 11200; }
  else               { t = 3; p = w - 6800; K = K15; k0 = keys.d0; k1 = keys.d1; idxOff = 13600; }
  int halfSid = 16 * K;                    // (K*32)/2 sids in half the counter range
  int sidA = p, sidB = p + halfSid;        // sid = ((kk*B+bb)*NCLS+n)
  int nA = sidA & 7, bbA = (sidA >> 3) & 3, kkA = sidA >> 5;
  int nB = sidB & 7, bbB = (sidB >> 3) & 3, kkB = sidB >> 5;
  const unsigned* wA = maskbits + ((t * NCLS + nA) * BATCH + bbA) * NWORDS;
  const unsigned* wB = maskbits + ((t * NCLS + nB) * BATCH + bbB) * NWORDS;
  unsigned baseA = ((unsigned)sidA) << 16, baseB = ((unsigned)sidB) << 16;

  unsigned long long bestA = 0ull, bestB = 0ull;
  for (int it = 0; it < NWORDS; ++it) {
    unsigned hw = ((unsigned)it << 5) + (unsigned)lane;
    unsigned o0, o1;
    threefry2x32(k0, k1, baseA + hw, baseB + hw, o0, o1);
    unsigned mA = (wA[it] >> lane) & 1u;
    unsigned mB = (wB[it] >> lane) & 1u;
    unsigned long long pA = ((unsigned long long)mA << 40)
                          | ((unsigned long long)(o0 >> 9) << 16)
                          | (unsigned long long)(65535u - hw);
    unsigned long long pB = ((unsigned long long)mB << 40)
                          | ((unsigned long long)(o1 >> 9) << 16)
                          | (unsigned long long)(65535u - hw);
    if (pA > bestA) bestA = pA;
    if (pB > bestB) bestB = pB;
  }
  // wave max reduce (32-bit shuffles for portability)
  for (int o = 16; o; o >>= 1) {
    unsigned hi = __shfl_xor((unsigned)(bestA >> 32), o, 32);
    unsigned lo = __shfl_xor((unsigned)bestA, o, 32);
    unsigned long long v = ((unsigned long long)hi << 32) | lo;
    if (v > bestA) bestA = v;
    hi = __shfl_xor((unsigned)(bestB >> 32), o, 32);
    lo = __shfl_xor((unsigned)bestB, o, 32);
    v = ((unsigned long long)hi << 32) | lo;
    if (v > bestB) bestB = v;
  }
  if (lane == 0) {
    idxArr[idxOff + (nA * BATCH + bbA) * K + kkA] = 65535 - (int)(bestA & 0xFFFFull);
    idxArr[idxOff + (nB * BATCH + bbB) * K + kkB] = 65535 - (int)(bestB & 0xFFFFull);
  }
}

// ---------------- kernel 5: gather + bilinear + L2-normalize -> f16 ----------------
__global__ void k_gather(const float* __restrict__ x, const int* __restrict__ idxArr,
                         _Float16* __restrict__ mTP, _Float16* __restrict__ mTN,
                         _Float16* __restrict__ mFP, _Float16* __restrict__ mFN) {
  int bid = blockIdx.x, c = threadIdx.x;
  int n, col, K, idxOff, qpad; _Float16* mat;
  if (bid < 5600)       {               n = bid / Q_BIG; col = bid % Q_BIG; K = K35; idxOff = 0;     mat = mTP; qpad = QP_BIG; }
  else if (bid < 11200) { bid -= 5600;  n = bid / Q_BIG; col = bid % Q_BIG; K = K35; idxOff = 5600;  mat = mTN; qpad = QP_BIG; }
  else if (bid < 13600) { bid -= 11200; n = bid / Q_SML; col = bid % Q_SML; K = K15; idxOff = 11200; mat = mFP; qpad = QP_SML; }
  else                  { bid -= 13600; n = bid / Q_SML; col = bid % Q_SML; K = K15; idxOff = 13600; mat = mFN; qpad = QP_SML; }
  int bb = col / K;
  int hw = idxArr[idxOff + n * BATCH * K + col];
  int y = hw >> 8, xc = hw & 255;
  const float SCL = 63.0f / 255.0f;                    // align_corners scale
  float py = y * SCL, px = xc * SCL;
  int y0 = (int)py, x0 = (int)px;
  float ty = py - (float)y0, tx = px - (float)x0;
  int y1 = y0 + 1 < 64 ? y0 + 1 : 63;
  int x1 = x0 + 1 < 64 ? x0 + 1 : 63;
  const float* xb = x + (size_t)(bb * CH + c) * HS * WSRC;
  float v00 = xb[y0 * WSRC + x0], v01 = xb[y0 * WSRC + x1];
  float v10 = xb[y1 * WSRC + x0], v11 = xb[y1 * WSRC + x1];
  float v = (v00 * (1.f - ty) + v10 * ty) * (1.f - tx)
          + (v01 * (1.f - ty) + v11 * ty) * tx;
  __shared__ float red[CH];
  red[c] = v * v; __syncthreads();
  for (int s = CH / 2; s; s >>= 1) { if (c < s) red[c] += red[c + s]; __syncthreads(); }
  float inv = 1.0f / (1e-6f + sqrtf(red[0]));
  mat[((size_t)n * qpad + col) * CH + c] = (_Float16)(v * inv);   // K-contiguous layout
}

// ---------------- kernel 6: einsum tiles via v_wmma_f32_16x16x32_f16 ----------------
// pair 0: FP*TP (-), 1: FP*TN (+), 2: FN*TP (+), 3: FN*TN (-)
__global__ __launch_bounds__(32) void k_mm(const _Float16* __restrict__ mTP,
                                           const _Float16* __restrict__ mTN,
                                           const _Float16* __restrict__ mFP,
                                           const _Float16* __restrict__ mFN,
                                           float* __restrict__ partial) {
  int bid  = blockIdx.x;
  int pair = bid / PER_PAIR;
  int r    = bid % PER_PAIR;
  int n    = r / (PT_TILES * QT_TILES);
  int r2   = r % (PT_TILES * QT_TILES);
  int pt   = r2 / QT_TILES, qt = r2 % QT_TILES;
  const _Float16* left  = ((pair < 2) ? mFP : mFN) + (size_t)n * QP_SML * CH;
  const _Float16* right = (((pair & 1) == 0) ? mTP : mTN) + (size_t)n * QP_BIG * CH;
  float sgn = (pair == 0 || pair == 3) ? -1.0f : 1.0f;

  int lane = threadIdx.x;
  int lm = lane & 15, hi = lane >> 4;
  const _Float16* arow = left  + (size_t)(pt * 16 + lm) * CH;   // A: M=sample p, K=channel
  const _Float16* brow = right + (size_t)(qt * 16 + lm) * CH;   // B: N=sample q, K=channel

  v8f acc = {};
  #pragma unroll
  for (int k0 = 0; k0 < CH; k0 += 32) {
    v16h a = *(const v16h*)(arow + k0 + hi * 16);
    v16h b = *(const v16h*)(brow + k0 + hi * 16);
    acc = __builtin_amdgcn_wmma_f32_16x16x32_f16(false, a, false, b, (short)0, acc,
                                                 false, false);
  }
  float s = acc[0]+acc[1]+acc[2]+acc[3]+acc[4]+acc[5]+acc[6]+acc[7];
  for (int o = 16; o; o >>= 1) s += __shfl_xor(s, o, 32);
  if (lane == 0) partial[bid] = sgn * s * (1.0f / 6720000.0f);  // /(300*700)/4/8
}

// ---------------- kernel 7: deterministic final reduce ----------------
__global__ void k_reduce(const float* __restrict__ partial, float* __restrict__ out) {
  __shared__ float red[256];
  int tid = threadIdx.x;
  float s = 0.f;
  for (int i = tid; i < NTILES; i += 256) s += partial[i];
  red[tid] = s; __syncthreads();
  for (int k = 128; k; k >>= 1) { if (tid < k) red[tid] += red[tid + k]; __syncthreads(); }
  if (tid == 0) out[0] = red[0];
}

// ---------------- launch ----------------
extern "C" void kernel_launch(void* const* d_in, const int* in_sizes, int n_in,
                              void* d_out, int out_size, void* d_ws, size_t ws_size,
                              hipStream_t stream) {
  const float* x     = (const float*)d_in[0];
  const float* logit = (const float*)d_in[1];
  const int*   label = (const int*)d_in[2];
  float* out = (float*)d_out;
  char* ws = (char*)d_ws;

  unsigned char* pred   = (unsigned char*)(ws + OFF_PRED);
  int*       idx        = (int*)(ws + OFF_IDX);
  unsigned*  maskbits   = (unsigned*)(ws + OFF_MASK);
  _Float16*  mTP        = (_Float16*)(ws + OFF_TP);
  _Float16*  mTN        = (_Float16*)(ws + OFF_TN);
  _Float16*  mFP        = (_Float16*)(ws + OFF_FP);
  _Float16*  mFN        = (_Float16*)(ws + OFF_FN);
  float*     partial    = (float*)(ws + OFF_PART);

  // jax: split(key(42), 4) == threefry(key=(0,42), counts=iota(8)) reshaped (4,2)
  unsigned xa[4], ya[4];
  for (unsigned j = 0; j < 4; ++j) threefry2x32(0u, 42u, j, j + 4u, xa[j], ya[j]);
  Keys4 keys = { xa[0], xa[1], xa[2], xa[3], ya[0], ya[1], ya[2], ya[3] };

  k_pred  <<<(BATCH * HW) / 256, 256, 0, stream>>>(logit, pred);
  k_mask  <<<1024, 256, 0, stream>>>(pred, label, maskbits);
  k_zero  <<<(unsigned)((ZERO_WORDS + 255) / 256), 256, 0, stream>>>(
            (unsigned*)(ws + OFF_TP), (unsigned long long)ZERO_WORDS);
  k_sample<<<1000, 256, 0, stream>>>(maskbits, keys, idx);
  k_gather<<<16000, 256, 0, stream>>>(x, idx, mTP, mTN, mFP, mFN);
  k_mm    <<<NTILES, 32, 0, stream>>>(mTP, mTN, mFP, mFN, partial);
  k_reduce<<<1, 256, 0, stream>>>(partial, out);
}
